// MultiHeadAttention_27410481283073
// MI455X (gfx1250) — matrix operand order repaired and checked by reading, not compiled, not run on hardware
//
#include <hip/hip_runtime.h>

// ---------------- problem constants ----------------
#define B_    2
#define S_    2048
#define H_    16
#define D_    2048
#define NOPE_ 64
#define ROPE_ 64
#define QK_   128
#define VD_   128
#define QR_   682
#define QRP_  704           // QR padded to multiple of 64 (and 32 for K)
#define KVR_  256
#define BS_   (B_ * S_)
#define KVB_N (H_ * (NOPE_ + VD_))   // 3072

#define WT_ROWS   64        // N rows of W staged per k-step
#define WT_K      32        // K per step
#define WT_STRIDE 40        // bf16 elems per LDS row: 32 data + 8 pad (80B pitch, conflict-free)

typedef __attribute__((ext_vector_type(16))) __bf16 bf16x16;
typedef __attribute__((ext_vector_type(8)))  __bf16 bf16x8;
typedef __attribute__((ext_vector_type(8)))  float  f32x8;
typedef __attribute__((ext_vector_type(4)))  unsigned int u32x4;
typedef __attribute__((ext_vector_type(8)))  int          i32x8;
typedef __attribute__((ext_vector_type(4)))  int          i32x4;

static __device__ __forceinline__ __bf16 to_bf16(float f) {
  unsigned u = __builtin_bit_cast(unsigned, f);
  u += 0x7FFFu + ((u >> 16) & 1u);              // round-to-nearest-even
  unsigned short h = (unsigned short)(u >> 16);
  return __builtin_bit_cast(__bf16, h);
}

static __device__ __forceinline__ bf16x16 pack16(const __bf16* p0, const __bf16* p1) {
  bf16x8 lo = *(const bf16x8*)p0;
  bf16x8 hi = *(const bf16x8*)p1;
  bf16x16 r;
#pragma unroll
  for (int i = 0; i < 8; ++i) { r[i] = lo[i]; r[8 + i] = hi[i]; }
  return r;
}

// Issue a TDM load of a WT_ROWS x WT_K bf16 tile of W into LDS (one per workgroup,
// issued by wave 0; tracked by TENSORcnt). D# layout per CDNA5 ISA ch.8:
//   group0: count=1 | lds_addr | global_addr | type=2
//   group1: data_size=2B, pad(16dw data + 4dw pad -> 80B pitch), dims/strides
static __device__ __forceinline__ void tdm_load_wtile(
    const __bf16* gptr, unsigned lds_byte_off, int N, int K) {
  unsigned long long ga = (unsigned long long)(uintptr_t)gptr;
  u32x4 g0;
  g0[0] = 1u;                                           // count=1, user descriptor
  g0[1] = lds_byte_off;                                 // lds_addr (bytes)
  g0[2] = (unsigned)(ga & 0xFFFFFFFFu);                 // global_addr[31:0]
  g0[3] = (unsigned)((ga >> 32) & 0x01FFFFFFu) | (2u << 30);  // addr[56:32] | type=2
  i32x8 g1;
  g1[0] = (1 << 16)                                     // data_size = 1 -> 2 bytes
        | (1 << 20)                                     // pad_enable
        | (3 << 22)                                     // pad_interval: 16 DWORDs
        | (3 << 25);                                    // pad_amount: 4 DWORDs
  g1[1] = (int)((unsigned)(K & 0xFFFF) << 16);          // tensor_dim0[15:0]
  g1[2] = (int)(((unsigned)K >> 16) & 0xFFFFu)          // tensor_dim0[31:16]
        | (int)((unsigned)(N & 0xFFFF) << 16);          // tensor_dim1[15:0]
  g1[3] = (int)(((unsigned)N >> 16) & 0xFFFFu)          // tensor_dim1[31:16]
        | (WT_K << 16);                                 // tile_dim0 = 32 elems
  g1[4] = WT_ROWS;                                      // tile_dim1 = 64 rows, tile_dim2 = 0
  g1[5] = K;                                            // tensor_dim0_stride[31:0] (elems)
  g1[6] = 0;                                            // stride hi / dim1_stride lo
  g1[7] = 0;
  i32x4 z4 = {0, 0, 0, 0};
#if __clang_major__ >= 23
  i32x8 z8 = {0, 0, 0, 0, 0, 0, 0, 0};
  __builtin_amdgcn_tensor_load_to_lds(g0, g1, z4, z4, z8, 0);
#else
  __builtin_amdgcn_tensor_load_to_lds(g0, g1, z4, z4, 0);
#endif
}

// ---------------- fp32 -> bf16 with 2D zero-padding ----------------
__global__ __launch_bounds__(256) void cvt_pad_bf16_kernel(
    const float* __restrict__ in, __bf16* __restrict__ out,
    int R, int Cin, int Cp, size_t total) {
  size_t i = (size_t)blockIdx.x * 256 + threadIdx.x;
  if (i >= total) return;
  int r = (int)(i / Cp);
  int c = (int)(i % Cp);
  float v = (r < R && c < Cin) ? in[(size_t)r * Cin + c] : 0.f;
  out[i] = to_bf16(v);
}

// ---------------- RMSNorm row kernel: f32 in -> bf16 out (padded) ----------------
__global__ __launch_bounds__(256) void rmsnorm_bf16_kernel(
    const float* __restrict__ in, const float* __restrict__ w,
    __bf16* __restrict__ out, int n, int inStride, int outStride) {
  __shared__ float red[256];
  const int row = blockIdx.x;
  const float* x = in + (size_t)row * inStride;
  float ss = 0.f;
  for (int c = threadIdx.x; c < n; c += 256) { float v = x[c]; ss += v * v; }
  red[threadIdx.x] = ss;
  __syncthreads();
  for (int o = 128; o > 0; o >>= 1) {
    if ((int)threadIdx.x < o) red[threadIdx.x] += red[threadIdx.x + o];
    __syncthreads();
  }
  const float rinv = rsqrtf(red[0] / (float)n + 1e-6f);
  __bf16* y = out + (size_t)row * outStride;
  for (int c = threadIdx.x; c < outStride; c += 256)
    y[c] = to_bf16(c < n ? x[c] * rinv * w[c] : 0.f);
}

// ---------------- bf16 WMMA GEMM with TDM-staged weight tiles ----------------
// C[M,N] = A[M,K] * W[N,K]^T.  grid = (M/64, N/64), block = 128 (4 waves).
// Each wave owns 16 M rows; the 64x32 W tile is shared via LDS, double-buffered
// by the Tensor Data Mover (wave 0 issues, TENSORcnt-tracked, pad -> 80B pitch).
__global__ __launch_bounds__(128) void gemm_bf16_wmma_tdm_kernel(
    const __bf16* __restrict__ A, const __bf16* __restrict__ W,
    float* __restrict__ C, int N, int K) {
  __shared__ __bf16 wtile[2][WT_ROWS * WT_STRIDE];
  const int lane = threadIdx.x & 31;
  const int wave = threadIdx.x >> 5;
  const int g  = lane >> 4;
  const int ln = lane & 15;
  const int mbase = (blockIdx.x << 6) + (wave << 4);
  const int nbase = blockIdx.y << 6;

  f32x8 acc[4];
#pragma unroll
  for (int t = 0; t < 4; ++t)
#pragma unroll
    for (int r = 0; r < 8; ++r) acc[t][r] = 0.f;

  const __bf16* wbase = W + (size_t)nbase * K;
  if (wave == 0) tdm_load_wtile(wbase, 0u, N, K);   // prefetch tile 0

  const __bf16* arow = A + (size_t)(mbase + ln) * K;
  int buf = 0;
  for (int k0 = 0; k0 < K; k0 += WT_K) {
    // A operand first: independent of the LDS tile, overlaps the TDM wait
    bf16x16 a = pack16(arow + k0 + g * 8, arow + k0 + 16 + g * 8);

    if (wave == 0) __builtin_amdgcn_s_wait_tensorcnt(0);  // current tile landed
    __syncthreads();                                      // publish to all waves
    if (wave == 0 && (k0 + WT_K) < K)                     // prefetch next tile
      tdm_load_wtile(wbase + k0 + WT_K,
                     (unsigned)((buf ^ 1) * WT_ROWS * WT_STRIDE * 2), N, K);

    // The TDM write to LDS is invisible to the compiler (the LDS address is
    // descriptor *data*, not a pointer operand). Escape the tile pointer into
    // an asm with a memory clobber so the ds_loads below are materialized.
    const __bf16* wt = wtile[buf];
    asm volatile("" : : "v"(wt) : "memory");

    // Load all four B operands first so the 8 ds_load_b128 issue as one clause
    // and DS latency overlaps WMMA issue (staggered dscnt waits).
    bf16x16 bmat[4];
#pragma unroll
    for (int t = 0; t < 4; ++t) {
      // B operand: lane ln = column N, K = g*8..+7 and 16+g*8..+7 in the LDS row
      const __bf16* wrow = wt + (t * 16 + ln) * WT_STRIDE + g * 8;
      bmat[t] = pack16(wrow, wrow + 16);
    }
#pragma unroll
    for (int t = 0; t < 4; ++t)
      acc[t] = __builtin_amdgcn_wmma_f32_16x16x32_bf16(
          false, a, false, bmat[t], (short)0, acc[t], false, false);
    buf ^= 1;
  }
#pragma unroll
  for (int t = 0; t < 4; ++t)
#pragma unroll
    for (int r = 0; r < 8; ++r)
      C[(size_t)(mbase + r + 8 * g) * N + nbase + t * 16 + ln] = acc[t][r];
}

// ---------------- RoPE on q, scale fold, rearrange to [B,H,S,128] bf16 ----------------
__global__ __launch_bounds__(256) void rope_q_kernel(
    const float* __restrict__ q, __bf16* __restrict__ qarr, float qscale) {
  size_t i = (size_t)blockIdx.x * 256 + threadIdx.x;   // [b][h][s][d]
  if (i >= (size_t)B_ * H_ * S_ * QK_) return;
  const int d = (int)(i & 127);
  const int s = (int)((i >> 7) & (S_ - 1));
  const int h = (int)((i >> 18) & (H_ - 1));
  const int b = (int)(i >> 22);
  const float* qr = q + (size_t)(b * S_ + s) * (H_ * QK_) + h * QK_;
  float v;
  if (d < NOPE_) {
    v = qr[d];
  } else {
    const int j  = d - NOPE_;
    const int jj = (j < 32) ? j : j - 32;
    const float inv = __expf(-(float)jj * (0.03125f * __logf(128000.0f)));
    const float ang = (float)s * inv;
    float sn, cs;
    __sincosf(ang, &sn, &cs);
    const float a  = (j < 32) ? qr[NOPE_ + 2 * jj] : qr[NOPE_ + 2 * jj + 1];
    const float bb = (j < 32) ? qr[NOPE_ + 2 * jj + 1] : qr[NOPE_ + 2 * jj];
    v = (j < 32) ? (a * cs - bb * sn) : (a * cs + bb * sn);
  }
  qarr[i] = to_bf16(v * qscale);
}

// ---------------- RoPE on k_pe (shared across heads) -> [B,S,64] bf16 ----------------
__global__ __launch_bounds__(256) void rope_k_kernel(
    const float* __restrict__ ckv, __bf16* __restrict__ kpe) {
  size_t i = (size_t)blockIdx.x * 256 + threadIdx.x;   // [b][s][j]
  if (i >= (size_t)B_ * S_ * ROPE_) return;
  const int j = (int)(i & 63);
  const int s = (int)((i >> 6) & (S_ - 1));
  const int b = (int)(i >> 17);
  const float* pr = ckv + (size_t)(b * S_ + s) * (KVR_ + ROPE_) + KVR_;
  const int jj = (j < 32) ? j : j - 32;
  const float inv = __expf(-(float)jj * (0.03125f * __logf(128000.0f)));
  const float ang = (float)s * inv;
  float sn, cs;
  __sincosf(ang, &sn, &cs);
  const float a  = (j < 32) ? pr[2 * jj] : pr[2 * jj + 1];
  const float bb = (j < 32) ? pr[2 * jj + 1] : pr[2 * jj];
  const float v  = (j < 32) ? (a * cs - bb * sn) : (a * cs + bb * sn);
  kpe[i] = to_bf16(v);
}

// ---------------- build key [B,H,S,128] and V^T [B,H,128,S] bf16 ----------------
__global__ __launch_bounds__(256) void kv_arrange_kernel(
    const float* __restrict__ kv, const __bf16* __restrict__ kpe,
    __bf16* __restrict__ karr, __bf16* __restrict__ vt) {
  size_t i = (size_t)blockIdx.x * 256 + threadIdx.x;   // [b][h][s][d]
  if (i >= (size_t)B_ * H_ * S_ * QK_) return;
  const int d = (int)(i & 127);
  const int s = (int)((i >> 7) & (S_ - 1));
  const int h = (int)((i >> 18) & (H_ - 1));
  const int b = (int)(i >> 22);
  const size_t row = (size_t)(b * S_ + s);
  __bf16 kval;
  if (d < NOPE_) kval = to_bf16(kv[row * KVB_N + h * (NOPE_ + VD_) + d]);
  else           kval = kpe[row * ROPE_ + (d - NOPE_)];
  karr[i] = kval;
  const float v = kv[row * KVB_N + h * (NOPE_ + VD_) + NOPE_ + d];
  vt[((size_t)(b * H_ + h) * VD_ + d) * S_ + s] = to_bf16(v);
}

// ---------------- flash attention, transposed-score WMMA ----------------
// Q pre-scaled by (1/sqrt(128))*log2(e); Kd row-major [B,H,S,128]; Vt [B,H,128,S].
// grid = (S/16, H, B), block = 1 wave.
__global__ __launch_bounds__(32) void mla_flash_attn_kernel(
    const __bf16* __restrict__ Q, const __bf16* __restrict__ Kd,
    const __bf16* __restrict__ Vt, __bf16* __restrict__ O) {
  const int lane = threadIdx.x;
  const int g  = lane >> 4;
  const int ln = lane & 15;
  const int q0 = blockIdx.x << 4;
  const int h  = blockIdx.y;
  const int b  = blockIdx.z;
  const size_t head = (size_t)(b * H_ + h) * S_ * QK_;

  // Q as the B operand of S^T = K * Q^T : lane ln = query column
  bf16x16 qb[4];
  {
    const __bf16* qr = Q + head + (size_t)(q0 + ln) * QK_;
#pragma unroll
    for (int t = 0; t < 4; ++t)
      qb[t] = pack16(qr + t * 32 + g * 8, qr + t * 32 + 16 + g * 8);
  }

  f32x8 acc[8];
#pragma unroll
  for (int j = 0; j < 8; ++j)
#pragma unroll
    for (int r = 0; r < 8; ++r) acc[j][r] = 0.f;
  float m = -3.0e38f, l = 0.f;

  for (int kt = 0; kt < S_; kt += 32) {
    f32x8 s0, s1;
#pragma unroll
    for (int r = 0; r < 8; ++r) { s0[r] = 0.f; s1[r] = 0.f; }
    const __bf16* kr0 = Kd + head + (size_t)(kt + ln) * QK_;
    const __bf16* kr1 = kr0 + 16 * QK_;
#pragma unroll
    for (int t = 0; t < 4; ++t) {
      bf16x16 a0 = pack16(kr0 + t * 32 + g * 8, kr0 + t * 32 + 16 + g * 8);
      s0 = __builtin_amdgcn_wmma_f32_16x16x32_bf16(false, a0, false, qb[t], (short)0, s0, false, false);
      bf16x16 a1 = pack16(kr1 + t * 32 + g * 8, kr1 + t * 32 + 16 + g * 8);
      s1 = __builtin_amdgcn_wmma_f32_16x16x32_bf16(false, a1, false, qb[t], (short)0, s1, false, false);
    }
    // online softmax (rows = queries live per-lane; halves combined via shfl_xor 16)
    float tm = s0[0];
#pragma unroll
    for (int r = 1; r < 8; ++r) tm = fmaxf(tm, s0[r]);
#pragma unroll
    for (int r = 0; r < 8; ++r) tm = fmaxf(tm, s1[r]);
    tm = fmaxf(tm, __shfl_xor(tm, 16, 32));
    const float mnew  = fmaxf(m, tm);
    const float alpha = exp2f(m - mnew);
    // P values land exactly in the A-operand layout for P*V (no lane movement)
    bf16x16 pa;
    float rs = 0.f;
#pragma unroll
    for (int r = 0; r < 8; ++r) { float p = exp2f(s0[r] - mnew); rs += p; pa[r] = to_bf16(p); }
#pragma unroll
    for (int r = 0; r < 8; ++r) { float p = exp2f(s1[r] - mnew); rs += p; pa[8 + r] = to_bf16(p); }
    rs += __shfl_xor(rs, 16, 32);
    l = l * alpha + rs;
    m = mnew;
    // rescale accumulators: row r+8g needs alpha of query (r+8g), stored at that lane
    float af[8];
#pragma unroll
    for (int r = 0; r < 8; ++r) af[r] = __shfl(alpha, r + (g << 3), 32);
#pragma unroll
    for (int j = 0; j < 8; ++j)
#pragma unroll
      for (int r = 0; r < 8; ++r) acc[j][r] *= af[r];
    // O += P * V  (V^T rows are contiguous keys -> clean B-operand loads)
#pragma unroll
    for (int j = 0; j < 8; ++j) {
      const __bf16* vp = Vt + head + (size_t)(j * 16 + ln) * S_ + kt + g * 8;
      bf16x16 vb = pack16(vp, vp + 16);
      acc[j] = __builtin_amdgcn_wmma_f32_16x16x32_bf16(false, pa, false, vb, (short)0, acc[j], false, false);
    }
  }
  const float linv = 1.f / l;
  float lf[8];
#pragma unroll
  for (int r = 0; r < 8; ++r) lf[r] = __shfl(linv, r + (g << 3), 32);
#pragma unroll
  for (int j = 0; j < 8; ++j)
#pragma unroll
    for (int r = 0; r < 8; ++r)
      O[(size_t)(b * S_ + q0 + r + 8 * g) * (H_ * VD_) + h * VD_ + j * 16 + ln] =
          to_bf16(acc[j][r] * lf[r]);
}

// ---------------- orchestration ----------------
extern "C" void kernel_launch(void* const* d_in, const int* in_sizes, int n_in,
                              void* d_out, int out_size, void* d_ws, size_t ws_size,
                              hipStream_t stream) {
  (void)in_sizes; (void)n_in; (void)out_size; (void)ws_size;
  const float* x       = (const float*)d_in[0];
  const float* q_a_w   = (const float*)d_in[1];
  const float* q_a_ln  = (const float*)d_in[2];
  const float* q_b_w   = (const float*)d_in[3];
  const float* kv_a_w  = (const float*)d_in[4];
  const float* kv_a_ln = (const float*)d_in[5];
  const float* kv_b_w  = (const float*)d_in[6];
  const float* o_w     = (const float*)d_in[7];
  float* out = (float*)d_out;

  char* ws = (char*)d_ws;
  size_t off = 0;
  auto alloc = [&](size_t bytes) -> char* {
    char* p = ws + off;
    off += (bytes + 255) & ~(size_t)255;
    return p;
  };

  __bf16* xbf     = (__bf16*)alloc((size_t)BS_ * D_ * 2);            // reused as qarr later
  __bf16* wqabf   = (__bf16*)alloc((size_t)QRP_ * D_ * 2);
  __bf16* wqbbf   = (__bf16*)alloc((size_t)(H_ * QK_) * QRP_ * 2);
  __bf16* wkvabf  = (__bf16*)alloc((size_t)(KVR_ + ROPE_) * D_ * 2);
  __bf16* wkvbbf  = (__bf16*)alloc((size_t)KVB_N * KVR_ * 2);
  __bf16* wobf    = (__bf16*)alloc((size_t)D_ * (H_ * VD_) * 2);
  float*  qlat    = (float*) alloc((size_t)BS_ * QRP_ * 4);
  __bf16* qlatn   = (__bf16*)alloc((size_t)BS_ * QRP_ * 2);
  float*  big     = (float*) alloc((size_t)BS_ * KVB_N * 4);         // q f32, then kv f32
  float*  ckv     = (float*) alloc((size_t)BS_ * (KVR_ + ROPE_) * 4);
  __bf16* cn      = (__bf16*)alloc((size_t)BS_ * KVR_ * 2);
  __bf16* kpe     = (__bf16*)alloc((size_t)BS_ * ROPE_ * 2);
  __bf16* karr    = (__bf16*)alloc((size_t)B_ * H_ * S_ * QK_ * 2);
  __bf16* vt      = (__bf16*)alloc((size_t)B_ * H_ * S_ * VD_ * 2);
  __bf16* attnout = (__bf16*)alloc((size_t)BS_ * H_ * VD_ * 2);
  __bf16* qarr    = xbf;   // x_bf16 is dead after the two down-proj GEMMs

  auto cvt = [&](const float* in, __bf16* o, int R, int Cin, int Rp, int Cp) {
    size_t tot = (size_t)Rp * Cp;
    cvt_pad_bf16_kernel<<<dim3((unsigned)((tot + 255) / 256)), 256, 0, stream>>>(
        in, o, R, Cin, Cp, tot);
  };
  auto gemm = [&](const __bf16* A, const __bf16* W, float* C, int M, int N, int K) {
    gemm_bf16_wmma_tdm_kernel<<<dim3(M / 64, N / 64), 128, 0, stream>>>(A, W, C, N, K);
  };

  // one-time fp32 -> bf16 conversions (with zero padding where needed)
  cvt(x,      xbf,    BS_,           D_,   BS_,           D_);
  cvt(q_a_w,  wqabf,  QR_,           D_,   QRP_,          D_);
  cvt(q_b_w,  wqbbf,  H_ * QK_,      QR_,  H_ * QK_,      QRP_);
  cvt(kv_a_w, wkvabf, KVR_ + ROPE_,  D_,   KVR_ + ROPE_,  D_);
  cvt(kv_b_w, wkvbbf, KVB_N,         KVR_, KVB_N,         KVR_);
  cvt(o_w,    wobf,   D_,            H_ * VD_, D_,        H_ * VD_);

  // q_lat = x @ q_a_w^T ; ckv = x @ kv_a_w^T
  gemm(xbf, wqabf, qlat, BS_, QRP_, D_);
  gemm(xbf, wkvabf, ckv, BS_, KVR_ + ROPE_, D_);

  // rmsnorm(q_lat) -> bf16 ; q = qlatn @ q_b_w^T (f32 into `big`)
  rmsnorm_bf16_kernel<<<BS_, 256, 0, stream>>>(qlat, q_a_ln, qlatn, QR_, QRP_, QRP_);
  gemm(qlatn, wqbbf, big, BS_, H_ * QK_, QRP_);

  // RoPE + scale-fold q -> [B,H,S,128] bf16 (reuses x_bf16 storage)
  const float qscale = 0.08838834764831843f * 1.4426950408889634f;  // 1/sqrt(128) * log2(e)
  rope_q_kernel<<<(B_ * H_ * S_ * QK_) / 256, 256, 0, stream>>>(big, qarr, qscale);

  // kv path: rmsnorm(c), rope(k_pe), kv = c_n @ kv_b_w^T (f32 into `big`, q consumed)
  rmsnorm_bf16_kernel<<<BS_, 256, 0, stream>>>(ckv, kv_a_ln, cn, KVR_, KVR_ + ROPE_, KVR_);
  rope_k_kernel<<<(B_ * S_ * ROPE_) / 256, 256, 0, stream>>>(ckv, kpe);
  gemm(cn, wkvbbf, big, BS_, KVB_N, KVR_);

  // assemble key [B,H,S,128] and V^T [B,H,128,S]
  kv_arrange_kernel<<<(B_ * H_ * S_ * QK_) / 256, 256, 0, stream>>>(big, kpe, karr, vt);

  // flash attention
  mla_flash_attn_kernel<<<dim3(S_ / 16, H_, B_), 32, 0, stream>>>(qarr, karr, vt, attnout);

  // final projection: out = attn @ o_w^T (f32 output)
  gemm(attnout, wobf, out, BS_, D_, D_);
}
